// Attention_33_22256520527997
// MI455X (gfx1250) — compile-verified, matrix operand order checked by reading, not run on hardware
//
#include <hip/hip_runtime.h>
#include <hip/hip_bf16.h>

// ---------------- types ----------------
typedef __attribute__((ext_vector_type(16))) __bf16         v16bf;
typedef __attribute__((ext_vector_type(8)))  float          v8f;
typedef __attribute__((ext_vector_type(2)))  float          v2f;
typedef __attribute__((ext_vector_type(8)))  unsigned short v8us;
typedef __attribute__((ext_vector_type(16))) unsigned short v16us;
typedef __attribute__((ext_vector_type(4)))  unsigned int   v4u;
typedef __attribute__((ext_vector_type(8)))  int            v8i;
typedef __attribute__((ext_vector_type(4)))  int            v4i;

#define TILES 9
#define BATCH 4
#define CH    128
#define C3    384
#define HH    128
#define NPX   16384   // 128*128 pixels per tile
#define HEADS 8
#define HD    16
#define HIMG  384
#define WIMG  384
#define CSTRIDE (HIMG*WIMG)   // channel stride in x / out

// ---------------- bf16 helpers ----------------
__device__ __forceinline__ unsigned short f2bf(float f) {
  unsigned int u = __builtin_bit_cast(unsigned int, f);
  u += 0x7fffu + ((u >> 16) & 1u);               // round-to-nearest-even
  return (unsigned short)(u >> 16);
}
__device__ __forceinline__ float bf2f(unsigned short h) {
  unsigned int u = ((unsigned int)h) << 16;
  return __builtin_bit_cast(float, u);
}
__device__ __forceinline__ v16bf make_frag(v8us lo, v8us hi) {
  v16us w;
  __builtin_memcpy(&w, &lo, 16);
  __builtin_memcpy(reinterpret_cast<char*>(&w) + 16, &hi, 16);
  return __builtin_bit_cast(v16bf, w);
}

// ---------------- k0: qkv_w fp32 -> bf16 ----------------
__global__ void k0_cvt_wq(const float* __restrict__ qkv_w,
                          unsigned short* __restrict__ wq) {
  int i = blockIdx.x * 256 + threadIdx.x;
  if (i < TILES * C3 * CH) wq[i] = f2bf(qkv_w[i]);
}

// ---------------- k1: LayerNorm + qkv 1x1 GEMM (WMMA bf16) ----------------
// grid(128 px-blocks, 36 tb), 256 threads (8 waves). Each block: 128 pixels,
// all 384 outputs. C[384 x 128px] = Wq[384 x 128] * Xn[128 x 128px].
__global__ __launch_bounds__(256)
void k1_ln_qkv(const float* __restrict__ x,
               const float* __restrict__ ln_w, const float* __restrict__ ln_b,
               const unsigned short* __restrict__ wq,
               const float* __restrict__ qkv_b,
               unsigned short* __restrict__ qkv0) {
  __shared__ __align__(16) unsigned short Xs[128 * 136]; // [px][c], bf16
  __shared__ float s_sum[256], s_sq[256];
  __shared__ float s_mu[128], s_rstd[128];

  const int tid  = threadIdx.x;
  const int lane = tid & 31;
  const int wave = tid >> 5;
  const int pxblk = blockIdx.x;             // h within tile
  const int tb = blockIdx.y;
  const int t = tb >> 2, b = tb & 3;
  const int ti = t / 3, tj = t % 3;

  const float* xb = x + ((size_t)(b * CH) * HIMG + (size_t)ti * HH + pxblk) * WIMG
                      + tj * HH;
  const int px = tid & 127;
  const int half = tid >> 7;

  // LN statistics: 2 threads per pixel, 64 channels each
  float s = 0.f, sq = 0.f;
  for (int c = half * 64; c < half * 64 + 64; ++c) {
    float v = xb[(size_t)c * CSTRIDE + px];
    s += v; sq += v * v;
  }
  s_sum[tid] = s; s_sq[tid] = sq;
  __syncthreads();
  if (half == 0) {
    float tot = s_sum[tid] + s_sum[tid + 128];
    float tq  = s_sq[tid]  + s_sq[tid + 128];
    float mu  = tot * (1.f / 128.f);
    float var = tq * (1.f / 128.f) - mu * mu;
    s_mu[px] = mu;
    s_rstd[px] = rsqrtf(var + 1e-5f);
  }
  __syncthreads();
  {
    float mu = s_mu[px], rs = s_rstd[px];
    for (int c = half * 64; c < half * 64 + 64; ++c) {
      float v = xb[(size_t)c * CSTRIDE + px];
      float y = (v - mu) * rs * ln_w[t * CH + c] + ln_b[t * CH + c];
      Xs[px * 136 + c] = f2bf(y);
    }
  }
  __syncthreads();

  // GEMM phase: wave = 16-pixel column tile
  const int col  = (wave << 4) + (lane & 15);
  const int ksel = lane >> 4;
  v16bf Bf[4];
#pragma unroll
  for (int kk = 0; kk < 4; ++kk) {
    const unsigned short* rp = &Xs[col * 136 + kk * 32 + ksel * 8];
    v8us lo = *reinterpret_cast<const v8us*>(rp);
    v8us hi = *reinterpret_cast<const v8us*>(rp + 16);
    Bf[kk] = make_frag(lo, hi);
  }
  const int nglobal = pxblk * 128 + col;
  for (int rt = 0; rt < 24; ++rt) {
    v8f acc = {};
    const int arow = rt * 16 + (lane & 15);
    const unsigned short* wrow = wq + (size_t)(t * C3 + arow) * CH;
#pragma unroll
    for (int kk = 0; kk < 4; ++kk) {
      const unsigned short* ap = wrow + kk * 32 + ksel * 8;
      v8us lo = *reinterpret_cast<const v8us*>(ap);
      v8us hi = *reinterpret_cast<const v8us*>(ap + 16);
      v16bf Af = make_frag(lo, hi);
      acc = __builtin_amdgcn_wmma_f32_16x16x32_bf16(false, Af, false, Bf[kk],
                                                    (short)0, acc, false, false);
    }
#pragma unroll
    for (int g = 0; g < 8; ++g) {
      int row = rt * 16 + g + ((lane >> 4) << 3);
      float v = acc[g] + qkv_b[t * C3 + row];
      qkv0[((size_t)tb * C3 + row) * NPX + nglobal] = f2bf(v);
    }
  }
}

// ---------------- k2: depthwise 3x3 via TDM plane staging ----------------
// grid(384 ch, 36 tb), 256 threads. One tensor_load_to_lds DMAs the whole
// 128x128 bf16 plane (32 KB) into LDS; stencil runs out of LDS.
__global__ __launch_bounds__(256)
void k2_dwconv(const unsigned short* __restrict__ qkv0,
               const float* __restrict__ dw_w, const float* __restrict__ dw_b,
               unsigned short* __restrict__ qkv1) {
  __shared__ __align__(16) unsigned short P[128 * 128]; // full plane, bf16
  const int tid = threadIdx.x;
  const int ch = blockIdx.x;   // 0..383
  const int tb = blockIdx.y;   // 0..35
  const int t = tb >> 2;
  const unsigned short* src = qkv0 + ((size_t)tb * C3 + ch) * NPX;

  if (tid < 32) {  // wave 0 issues the TDM descriptor
    unsigned long long ga = (unsigned long long)(const void*)src;  // 57-bit VA
    unsigned int lds = (unsigned int)(unsigned long long)(void*)&P[0];
    v4u g0;
    g0[0] = 1u;                                     // count=1, user descriptor
    g0[1] = lds;                                    // lds_addr
    g0[2] = (unsigned int)(ga & 0xffffffffu);       // global_addr[31:0]
    g0[3] = (unsigned int)((ga >> 32) & 0x1ffffffu) // global_addr[56:32]
          | (2u << 30);                             // type=2 (image)
    v8i g1;
    g1[0] = (int)(1u << 16);      // data_size=1 (2 bytes/elem)
    g1[1] = (int)(128u << 16);    // tensor_dim0[15:0]=128 (bits 63:48)
    g1[2] = (int)(128u << 16);    // tensor_dim1[15:0]=128 (bits 95:80)
    g1[3] = (int)(128u << 16);    // tile_dim0=128 (bits 127:112)
    g1[4] = 128;                  // tile_dim1=128 (bits 143:128)
    g1[5] = 128;                  // tensor_dim0_stride[31:0]=128
    g1[6] = 0;
    g1[7] = 0;
    v4i gz = {0, 0, 0, 0};
#if defined(__clang_major__) && (__clang_major__ >= 23)
    v8i gz8 = {0, 0, 0, 0, 0, 0, 0, 0};
    __builtin_amdgcn_tensor_load_to_lds(g0, g1, gz, gz, gz8, 0);
#else
    __builtin_amdgcn_tensor_load_to_lds(g0, g1, gz, gz, 0);
#endif
    __builtin_amdgcn_s_wait_tensorcnt(0);
  }
  __syncthreads();

  const float* kwp = dw_w + (size_t)(t * C3 + ch) * 9;
  const float k00 = kwp[0], k01 = kwp[1], k02 = kwp[2];
  const float k10 = kwp[3], k11 = kwp[4], k12 = kwp[5];
  const float k20 = kwp[6], k21 = kwp[7], k22 = kwp[8];
  const float bias = dw_b[t * C3 + ch];
  unsigned short* dst = qkv1 + ((size_t)tb * C3 + ch) * NPX;

  for (int i = 0; i < 64; ++i) {
    const int n = tid + i * 256;
    const int h = n >> 7, w = n & 127;
    float acc = bias;
    if (h > 0) {
      const int r = (h - 1) * 128 + w;
      if (w > 0)   acc += bf2f(P[r - 1]) * k00;
                   acc += bf2f(P[r])     * k01;
      if (w < 127) acc += bf2f(P[r + 1]) * k02;
    }
    {
      const int r = h * 128 + w;
      if (w > 0)   acc += bf2f(P[r - 1]) * k10;
                   acc += bf2f(P[r])     * k11;
      if (w < 127) acc += bf2f(P[r + 1]) * k12;
    }
    if (h < 127) {
      const int r = (h + 1) * 128 + w;
      if (w > 0)   acc += bf2f(P[r - 1]) * k20;
                   acc += bf2f(P[r])     * k21;
      if (w < 127) acc += bf2f(P[r + 1]) * k22;
    }
    dst[n] = f2bf(acc);
  }
}

// ---------------- k3: Gram(q,k) over N=16384 + softmax (WMMA f32) --------
// grid(8 heads, 36 tb), 128 threads (4 waves). attn: [tb][head][16][16] f32
__global__ __launch_bounds__(128)
void k3_attn(const unsigned short* __restrict__ qkv1,
             const float* __restrict__ temperature,
             float* __restrict__ attn) {
  __shared__ float Gram[4][16][16];
  __shared__ float sqp[4][32], skp[4][32];
  const int tid = threadIdx.x, lane = tid & 31, wave = tid >> 5;
  const int head = blockIdx.x, tb = blockIdx.y, t = tb >> 2;

  const unsigned short* qb = qkv1 + ((size_t)tb * C3 + head * HD) * NPX;
  const unsigned short* kb = qkv1 + ((size_t)tb * C3 + CH + head * HD) * NPX;
  const int row  = lane & 15;
  const int noff = (lane >> 4) * 2;
  v8f acc = {};
  float pq = 0.f, pk = 0.f;
  const size_t roff = (size_t)row * NPX;
  for (int n0 = wave * 4096; n0 < wave * 4096 + 4096; n0 += 4) {
    unsigned int qw = *reinterpret_cast<const unsigned int*>(qb + roff + n0 + noff);
    unsigned int kw = *reinterpret_cast<const unsigned int*>(kb + roff + n0 + noff);
    v2f a;   a.x   = bf2f((unsigned short)(qw & 0xffffu));
             a.y   = bf2f((unsigned short)(qw >> 16));
    v2f bb;  bb.x  = bf2f((unsigned short)(kw & 0xffffu));
             bb.y  = bf2f((unsigned short)(kw >> 16));
    pq += a.x * a.x + a.y * a.y;
    pk += bb.x * bb.x + bb.y * bb.y;
    acc = __builtin_amdgcn_wmma_f32_16x16x4_f32(false, a, false, bb,
                                                (short)0, acc, false, false);
  }
  sqp[wave][lane] = pq;
  skp[wave][lane] = pk;
#pragma unroll
  for (int g = 0; g < 8; ++g) {
    int r = g + ((lane >> 4) << 3);
    Gram[wave][r][lane & 15] = acc[g];
  }
  __syncthreads();

  if (tid < 16) {
    const int r = tid;
    float sq = 0.f;
    for (int w2 = 0; w2 < 4; ++w2) sq += sqp[w2][r] + sqp[w2][r + 16];
    float invq = 1.f / fmaxf(sqrtf(sq), 1e-12f);
    float temp = temperature[t * HEADS + head];
    float vals[16], mx = -3.4e38f;
    for (int c = 0; c < 16; ++c) {
      float g = Gram[0][r][c] + Gram[1][r][c] + Gram[2][r][c] + Gram[3][r][c];
      float sk = 0.f;
      for (int w2 = 0; w2 < 4; ++w2) sk += skp[w2][c] + skp[w2][c + 16];
      float invk = 1.f / fmaxf(sqrtf(sk), 1e-12f);
      float v = g * invq * invk * temp;
      vals[c] = v; mx = fmaxf(mx, v);
    }
    float sum = 0.f;
    for (int c = 0; c < 16; ++c) { vals[c] = __expf(vals[c] - mx); sum += vals[c]; }
    float inv = 1.f / sum;
    float* ab = attn + (size_t)(tb * HEADS + head) * 256 + r * 16;
    for (int c = 0; c < 16; ++c) ab[c] = vals[c] * inv;
  }
}

// ---------------- k4: M = proj_w @ blockdiag(attn)  (WMMA f32) -----------
// grid(36 tb), 128 threads (4 waves), 64 16x16 output tiles.
__global__ __launch_bounds__(128)
void k4_projM(const float* __restrict__ proj_w,
              const float* __restrict__ attn,
              unsigned short* __restrict__ Mw) {
  const int tid = threadIdx.x, lane = tid & 31, wave = tid >> 5;
  const int tb = blockIdx.x, t = tb >> 2;
  for (int i = 0; i < 16; ++i) {
    const int id = wave * 16 + i;
    const int head = id >> 3, r = id & 7;
    const float* ab = attn + (size_t)(tb * HEADS + head) * 256;
    v8f acc = {};
#pragma unroll
    for (int k0 = 0; k0 < 16; k0 += 4) {
      const int koff = k0 + ((lane >> 4) << 1);
      const int arow = r * 16 + (lane & 15);
      v2f a = *reinterpret_cast<const v2f*>(
          proj_w + (size_t)(t * CH + arow) * CH + head * HD + koff);
      v2f bb;
      bb.x = ab[koff * 16 + (lane & 15)];
      bb.y = ab[(koff + 1) * 16 + (lane & 15)];
      acc = __builtin_amdgcn_wmma_f32_16x16x4_f32(false, a, false, bb,
                                                  (short)0, acc, false, false);
    }
#pragma unroll
    for (int g = 0; g < 8; ++g) {
      int ro = r * 16 + g + ((lane >> 4) << 3);
      int co = head * HD + (lane & 15);
      Mw[((size_t)tb * CH + ro) * CH + co] = f2bf(acc[g]);
    }
  }
}

// ---------------- k5: out = M @ v + proj_b + grw * res  (WMMA bf16) ------
// grid(128 px-blocks, 36 tb), 256 threads (8 waves).
__global__ __launch_bounds__(256)
void k5_out(const unsigned short* __restrict__ qkv1,
            const unsigned short* __restrict__ Mw,
            const float* __restrict__ proj_b,
            const float* __restrict__ grw,
            const float* __restrict__ x,
            float* __restrict__ out) {
  __shared__ __align__(16) unsigned short Vs[128 * 136]; // [px][ch], bf16
  const int tid = threadIdx.x, lane = tid & 31, wave = tid >> 5;
  const int pxblk = blockIdx.x, tb = blockIdx.y;
  const int t = tb >> 2, b = tb & 3;
  const int ti = t / 3, tj = t % 3;

  const int col  = (wave << 4) + (lane & 15);
  const size_t base = ((size_t)(b * CH) * HIMG + (size_t)ti * HH + pxblk) * WIMG
                    + tj * HH + col;
  const float* xres = x + base;
  __builtin_prefetch(xres, 0, 0);   // global_prefetch for residual stream

  const unsigned short* vb = qkv1 + ((size_t)tb * C3 + 2 * CH) * NPX + pxblk * 128;
  for (int i = 0; i < 64; ++i) {
    int idx = tid + i * 256;
    int ch = idx >> 7, px = idx & 127;
    Vs[px * 136 + ch] = vb[(size_t)ch * NPX + px];   // transpose into LDS
  }
  __syncthreads();

  const int ksel = lane >> 4;
  v16bf Bf[4];
#pragma unroll
  for (int kk = 0; kk < 4; ++kk) {
    const unsigned short* rp = &Vs[col * 136 + kk * 32 + ksel * 8];
    v8us lo = *reinterpret_cast<const v8us*>(rp);
    v8us hi = *reinterpret_cast<const v8us*>(rp + 16);
    Bf[kk] = make_frag(lo, hi);
  }
  const float g = grw[t];
  float* ob = out + base;
  for (int rt = 0; rt < 8; ++rt) {
    v8f acc = {};
    const int arow = rt * 16 + (lane & 15);
    const unsigned short* mrow = Mw + ((size_t)tb * CH + arow) * CH;
#pragma unroll
    for (int kk = 0; kk < 4; ++kk) {
      const unsigned short* ap = mrow + kk * 32 + ksel * 8;
      v8us lo = *reinterpret_cast<const v8us*>(ap);
      v8us hi = *reinterpret_cast<const v8us*>(ap + 16);
      v16bf Af = make_frag(lo, hi);
      acc = __builtin_amdgcn_wmma_f32_16x16x32_bf16(false, Af, false, Bf[kk],
                                                    (short)0, acc, false, false);
    }
#pragma unroll
    for (int gg = 0; gg < 8; ++gg) {
      int row = rt * 16 + gg + ((lane >> 4) << 3);
      float v = acc[gg] + proj_b[t * CH + row] + g * xres[(size_t)row * CSTRIDE];
      ob[(size_t)row * CSTRIDE] = v;
    }
  }
}

// ---------------- host launch ----------------
extern "C" void kernel_launch(void* const* d_in, const int* in_sizes, int n_in,
                              void* d_out, int out_size, void* d_ws, size_t ws_size,
                              hipStream_t stream) {
  (void)in_sizes; (void)n_in; (void)out_size; (void)ws_size;
  const float* x      = (const float*)d_in[0];
  const float* ln_w   = (const float*)d_in[1];
  const float* ln_b   = (const float*)d_in[2];
  const float* qkv_w  = (const float*)d_in[3];
  const float* qkv_b  = (const float*)d_in[4];
  const float* dw_w   = (const float*)d_in[5];
  const float* dw_b   = (const float*)d_in[6];
  const float* proj_w = (const float*)d_in[7];
  const float* proj_b = (const float*)d_in[8];
  const float* temp   = (const float*)d_in[9];
  const float* grw    = (const float*)d_in[10];
  float* out = (float*)d_out;

  char* ws = (char*)d_ws;
  const size_t SZ_QKV = (size_t)TILES * BATCH * C3 * NPX * sizeof(unsigned short);
  const size_t SZ_ATT = (size_t)TILES * BATCH * HEADS * 256 * sizeof(float);
  unsigned short* wq   = (unsigned short*)(ws);
  unsigned short* qkv0 = (unsigned short*)(ws + (1u << 20));
  unsigned short* qkv1 = (unsigned short*)(ws + (1u << 20) + SZ_QKV);
  float*          attn = (float*)(ws + (1u << 20) + 2 * SZ_QKV);
  unsigned short* Mw   = (unsigned short*)(ws + (1u << 20) + 2 * SZ_QKV + SZ_ATT);

  k0_cvt_wq<<<(TILES * C3 * CH + 255) / 256, 256, 0, stream>>>(qkv_w, wq);
  k1_ln_qkv<<<dim3(128, 36), 256, 0, stream>>>(x, ln_w, ln_b, wq, qkv_b, qkv0);
  k2_dwconv<<<dim3(384, 36), 256, 0, stream>>>(qkv0, dw_w, dw_b, qkv1);
  k3_attn<<<dim3(8, 36), 128, 0, stream>>>(qkv1, temp, attn);
  k4_projM<<<36, 128, 0, stream>>>(proj_w, attn, Mw);
  k5_out<<<dim3(128, 36), 256, 0, stream>>>(qkv1, Mw, proj_b, grw, x, out);
}
